// T5Attention_48687749267902
// MI455X (gfx1250) — compile-verified, matrix operand order checked by reading, not run on hardware
//
#include <hip/hip_runtime.h>
#include <hip/hip_bf16.h>

// ---------------------------------------------------------------------------
// Types for CDNA5 WMMA (wave32): v_wmma_f32_16x16x32_bf16
// ---------------------------------------------------------------------------
typedef __attribute__((ext_vector_type(16))) __bf16 v16bf;
typedef __attribute__((ext_vector_type(8)))  float  v8f;

// Problem constants
#define BB 2
#define SS 2048
#define DD 1024
#define HH 16
#define DK 64
#define MROWS (BB * SS)        // 4096
#define NBIAS 4095             // distances in [-(S-1), S-1]

// ---------------------------------------------------------------------------
// Helpers
// ---------------------------------------------------------------------------
__device__ __forceinline__ unsigned short f2bf(float f) {
  unsigned int u = __float_as_uint(f);
  u += 0x7FFFu + ((u >> 16) & 1u);   // round-to-nearest-even
  return (unsigned short)(u >> 16);
}

// Build a v16bf A/B operand from two 16-byte chunks (ISA lane layout:
// elements 0..7 at k-base, elements 8..15 at k-base+16).
__device__ __forceinline__ v16bf ld16x2(const unsigned short* p0,
                                        const unsigned short* p1) {
  uint4 a = *(const uint4*)(const void*)p0;
  uint4 b = *(const uint4*)(const void*)p1;
  v16bf r;
  __builtin_memcpy(&r, &a, 16);
  __builtin_memcpy(reinterpret_cast<char*>(&r) + 16, &b, 16);
  return r;
}

__device__ __forceinline__ v8f wmma_bf16(v16bf a, v16bf b, v8f c) {
  return __builtin_amdgcn_wmma_f32_16x16x32_bf16(
      /*neg_a=*/false, a, /*neg_b=*/false, b,
      /*c_mod=*/(short)0, c, /*reuse_a=*/false, /*reuse_b=*/false);
}

// ---- CDNA5 async global->LDS copy (ASYNCcnt path) -------------------------
// Copies 16 bytes per lane from global to LDS without touching VGPR data.
// Inline asm (VGLOBAL async encoding): vdst = per-lane LDS byte offset (the
// LDS aperture truncates flat addresses to addr[31:0]), vaddr = 64-bit global
// address pair, saddr = off.
__device__ __forceinline__ void async_ld16(unsigned short* lds,
                                           const unsigned short* g) {
  unsigned int laddr = (unsigned int)(unsigned long long)lds;   // LDS offset
  unsigned long long gaddr = (unsigned long long)g;
  asm volatile("global_load_async_to_lds_b128 %0, %1, off"
               :: "v"(laddr), "v"(gaddr) : "memory");
}

__device__ __forceinline__ void wait_async0() {
#if __has_builtin(__builtin_amdgcn_s_wait_asynccnt)
  __builtin_amdgcn_s_wait_asynccnt(0);
#else
  asm volatile("s_wait_asynccnt 0x0" ::: "memory");
#endif
}

// ---------------------------------------------------------------------------
// fp32 -> bf16 convert (streaming)
// ---------------------------------------------------------------------------
__global__ void cvt_bf16_kernel(const float* __restrict__ in,
                                unsigned short* __restrict__ out, int n) {
  int i = blockIdx.x * 256 + threadIdx.x;
  if (i < n) out[i] = f2bf(in[i]);
}

// fp32 [K=1024][N=1024] -> bf16 [N][K] (transpose so B-operand columns are
// contiguous rows for WMMA)
__global__ void transpose_cvt_kernel(const float* __restrict__ in,
                                     unsigned short* __restrict__ out) {
  int i = blockIdx.x * 256 + threadIdx.x;   // 1M elements
  int n = i >> 10, k = i & 1023;
  out[n * 1024 + k] = f2bf(in[k * 1024 + n]);
}

// ---------------------------------------------------------------------------
// Relative-position bias table: biasTab[h][d+2047] = rel_table[bucket(d)][h]
// ---------------------------------------------------------------------------
__global__ void biastab_kernel(const float* __restrict__ rel_table,
                               float* __restrict__ biasTab) {
  int idx = blockIdx.x * 256 + threadIdx.x;
  if (idx >= HH * NBIAS) return;
  int h = idx / NBIAS;
  int d = idx % NBIAS - (SS - 1);   // rel_pos = k - q
  int n = -d;                       // bidirectional T5 bucket
  int ret = 0;
  if (n < 0) { ret = 16; n = -n; }
  int v;
  if (n < 8) {
    v = n;
  } else {
    // 8 + int(log(n/8)/log(16) * 8), clamped to 15
    float t = __logf((float)n * 0.125f) * (8.0f / __logf(16.0f));
    v = 8 + (int)t;
    if (v > 15) v = 15;
  }
  biasTab[idx] = rel_table[(v + ret) * HH + h];
}

// Stream the 268MB position_bias output (this dominates the roofline).
__global__ void posbias_kernel(const float* __restrict__ biasTab,
                               float* __restrict__ out) {
  int tid = blockIdx.x * 256 + threadIdx.x;     // groups of 4 along k
  if (tid >= (HH * SS * SS) / 4) return;
  int j = tid * 4;
  int h = j >> 22;                 // / (2048*2048)
  int rem = j & ((SS * SS) - 1);
  int q = rem >> 11;
  int k = rem & (SS - 1);
  int base = h * NBIAS + (k - q + (SS - 1));
  float4 v;
  v.x = biasTab[base + 0];
  v.y = biasTab[base + 1];
  v.z = biasTab[base + 2];
  v.w = biasTab[base + 3];
  *(float4*)&out[j] = v;
}

// ---------------------------------------------------------------------------
// bf16 GEMM: C[4096 x 1024] = A[4096 x 1024] * B (B given transposed [N][K]).
// One wave computes a 16x64 tile. k-loop unrolled by 2 with ping-pong operand
// buffers: no register rotation movs, loads for the next k-step in flight
// while the current WMMAs execute.
// mode 0/1: write bf16 as Q/K [b][h][s][d]
// mode 2:   write bf16 as Vt  [b][h][d][s]
// mode 3:   write f32 row-major [row][n]  (final output projection)
// ---------------------------------------------------------------------------
__global__ __launch_bounds__(128) void gemm_bf16_kernel(
    const unsigned short* __restrict__ A, const unsigned short* __restrict__ Bw,
    void* __restrict__ outp, int mode) {
  const int lane = threadIdx.x & 31;
  const int wave = threadIdx.x >> 5;
  const int wg   = blockIdx.x * 4 + wave;   // 0..4095
  const int m0   = (wg >> 4) * 16;          // row tile
  const int n0   = (wg & 15) * 64;          // col group
  const int half = lane >> 4;
  const int l16  = lane & 15;
  const int kb0  = half * 8;

  v8f acc[4] = {};
  const unsigned short* Abase = A + (m0 + l16) * 1024;
  const unsigned short* Bbase[4];
#pragma unroll
  for (int g = 0; g < 4; ++g) Bbase[g] = Bw + (n0 + g * 16 + l16) * 1024;

  // prologue: buffer 0 holds k-step 0
  v16bf a0 = ld16x2(Abase + kb0, Abase + kb0 + 16);
  v16bf b0[4], a1, b1[4];
#pragma unroll
  for (int g = 0; g < 4; ++g)
    b0[g] = ld16x2(Bbase[g] + kb0, Bbase[g] + kb0 + 16);

  for (int k0 = 0; k0 < 1024; k0 += 64) {
    {   // prefetch k0+32 into buffer 1 (always in range: k0 <= 960)
      int kb = k0 + 32 + kb0;
      a1 = ld16x2(Abase + kb, Abase + kb + 16);
#pragma unroll
      for (int g = 0; g < 4; ++g)
        b1[g] = ld16x2(Bbase[g] + kb, Bbase[g] + kb + 16);
    }
#pragma unroll
    for (int g = 0; g < 4; ++g) acc[g] = wmma_bf16(a0, b0[g], acc[g]);
    if (k0 + 64 < 1024) {   // prefetch k0+64 into buffer 0
      int kb = k0 + 64 + kb0;
      a0 = ld16x2(Abase + kb, Abase + kb + 16);
#pragma unroll
      for (int g = 0; g < 4; ++g)
        b0[g] = ld16x2(Bbase[g] + kb, Bbase[g] + kb + 16);
    }
#pragma unroll
    for (int g = 0; g < 4; ++g) acc[g] = wmma_bf16(a1, b1[g], acc[g]);
  }

#pragma unroll
  for (int g = 0; g < 4; ++g) {
#pragma unroll
    for (int r = 0; r < 8; ++r) {
      int row = m0 + r + half * 8;        // flattened b*S+s
      int n   = n0 + g * 16 + l16;        // 0..1023
      float v = acc[g][r];
      if (mode == 3) {
        ((float*)outp)[row * 1024 + n] = v;
      } else {
        int b2 = row >> 11, s = row & (SS - 1);
        int h = n >> 6,     d = n & 63;
        unsigned short bv = f2bf(v);
        if (mode == 2)
          ((unsigned short*)outp)[((b2 * HH + h) * DK + d) * SS + s] = bv;
        else
          ((unsigned short*)outp)[((b2 * HH + h) * SS + s) * DK + d] = bv;
      }
    }
  }
}

// ---------------------------------------------------------------------------
// Flash attention. One wave owns a 16-row q-tile; loop kv in chunks of 32.
// All 4 waves of a block share each K/V chunk, so chunks are staged into LDS
// with CDNA5 async global->LDS copies, double-buffered: the copy of chunk i+1
// overlaps compute of chunk i. scores via 2x wmma (K-dim = Dk = 64), online
// softmax, P staged through LDS into A-operand layout, P.V via 4x wmma.
// ---------------------------------------------------------------------------
__global__ __launch_bounds__(128) void attn_kernel(
    const unsigned short* __restrict__ Qb, const unsigned short* __restrict__ Kb,
    const unsigned short* __restrict__ Vt, const float* __restrict__ biasTab,
    const float* __restrict__ mask, unsigned short* __restrict__ attnout) {
  __shared__ float sbias[NBIAS];
  __shared__ __align__(16) unsigned short sp[4][16 * 32];
  __shared__ __align__(16) unsigned short sK[2][32 * 64];   // [kv][d]
  __shared__ __align__(16) unsigned short sV[2][64 * 32];   // [d][kv]

  const int tid  = threadIdx.x;
  const int lane = tid & 31;
  const int wave = tid >> 5;
  const int bh   = blockIdx.x >> 5;     // b*16 + h
  const int qb   = blockIdx.x & 31;
  const int b    = bh >> 4;
  const int h    = bh & 15;
  const int q0   = qb * 64 + wave * 16;
  const int half = lane >> 4;
  const int l16  = lane & 15;
  const int kb0  = half * 8;

  const unsigned short* Kbase = Kb + bh * SS * DK;   // [s][d]
  const unsigned short* Vbase = Vt + bh * DK * SS;   // [d][s]

  for (int i = tid; i < NBIAS; i += 128)
    sbias[i] = biasTab[h * NBIAS + i];

  // Q rows for this lane, A-operand chunks over d (two k-slices of 32)
  const unsigned short* Qrow = Qb + (bh * SS + q0 + l16) * DK;
  v16bf Aq0 = ld16x2(Qrow + kb0,      Qrow + kb0 + 16);
  v16bf Aq1 = ld16x2(Qrow + 32 + kb0, Qrow + 48 + kb0);

  float m[8], l[8];
  v8f acc[4] = {};
#pragma unroll
  for (int r = 0; r < 8; ++r) { m[r] = -3.0e38f; l[r] = 0.0f; }

  // --- issue async stage of chunk 0 (K: 32x64 bf16, V: 64x32 bf16) ---
#pragma unroll
  for (int c = tid; c < 256; c += 128) {            // K: 256 x 16B
    int row = c >> 3, col8 = (c & 7) * 8;
    async_ld16(&sK[0][row * 64 + col8], Kbase + row * 64 + col8);
  }
#pragma unroll
  for (int c = tid; c < 256; c += 128) {            // V: 256 x 16B
    int row = c >> 2, col8 = (c & 3) * 8;
    async_ld16(&sV[0][row * 32 + col8], Vbase + row * SS + col8);
  }

  for (int kv0 = 0; kv0 < SS; kv0 += 32) {
    const int buf = (kv0 >> 5) & 1;
    wait_async0();        // this wave's batch for `buf` complete
    __syncthreads();      // all waves' batches complete; prior reads of buf^1 done

    // --- overlap: issue async stage of next chunk into the other buffer ---
    if (kv0 + 32 < SS) {
      const int nb = buf ^ 1;
      const unsigned short* Kn = Kbase + (kv0 + 32) * 64;
#pragma unroll
      for (int c = tid; c < 256; c += 128) {
        int row = c >> 3, col8 = (c & 7) * 8;
        async_ld16(&sK[nb][row * 64 + col8], Kn + row * 64 + col8);
      }
#pragma unroll
      for (int c = tid; c < 256; c += 128) {
        int row = c >> 2, col8 = (c & 3) * 8;
        async_ld16(&sV[nb][row * 32 + col8], Vbase + row * SS + kv0 + 32 + col8);
      }
    }

    v8f st[2];
    float rmax[8];
#pragma unroll
    for (int r = 0; r < 8; ++r) rmax[r] = -3.0e38f;

    // ---- scores: two 16x16 tiles (columns kv0..kv0+31), K from LDS ----
#pragma unroll
    for (int t = 0; t < 2; ++t) {
      const unsigned short* Krow = &sK[buf][(t * 16 + l16) * 64];
      v16bf B0 = ld16x2(Krow + kb0,      Krow + kb0 + 16);
      v16bf B1 = ld16x2(Krow + 32 + kb0, Krow + 48 + kb0);
      v8f c = {};
      c = wmma_bf16(Aq0, B0, c);
      c = wmma_bf16(Aq1, B1, c);
      int kvp = kv0 + t * 16 + l16;               // this lane's column
      float mv = mask[b * SS + kvp];
#pragma unroll
      for (int r = 0; r < 8; ++r) {
        int qp = q0 + r + half * 8;
        c[r] += sbias[kvp - qp + (SS - 1)] + mv;
        rmax[r] = fmaxf(rmax[r], c[r]);
      }
      st[t] = c;
    }

    // ---- online softmax update ----
#pragma unroll
    for (int r = 0; r < 8; ++r) {
      float v = rmax[r];
#pragma unroll
      for (int off = 1; off < 16; off <<= 1)
        v = fmaxf(v, __shfl_xor(v, off, 32));
      float nm = fmaxf(m[r], v);
      float sc = __expf(m[r] - nm);
      m[r] = nm;
      float ps = 0.0f;
#pragma unroll
      for (int t = 0; t < 2; ++t) {
        st[t][r] = __expf(st[t][r] - nm);
        ps += st[t][r];
      }
#pragma unroll
      for (int off = 1; off < 16; off <<= 1)
        ps += __shfl_xor(ps, off, 32);
      l[r] = l[r] * sc + ps;
#pragma unroll
      for (int g = 0; g < 4; ++g) acc[g][r] *= sc;
    }

    // ---- stage P (C-layout) -> LDS row-major 16x32 bf16 ----
    __syncthreads();
#pragma unroll
    for (int t = 0; t < 2; ++t)
#pragma unroll
      for (int r = 0; r < 8; ++r)
        sp[wave][(r + half * 8) * 32 + t * 16 + l16] = f2bf(st[t][r]);
    __syncthreads();

    // ---- P.V: A-operand from LDS, B-operand from staged V ----
    const unsigned short* Pr = &sp[wave][l16 * 32 + kb0];
    v16bf Ap = ld16x2(Pr, Pr + 16);
#pragma unroll
    for (int g = 0; g < 4; ++g) {
      const unsigned short* Vrow = &sV[buf][(g * 16 + l16) * 32];
      v16bf Bv = ld16x2(Vrow + kb0, Vrow + kb0 + 16);
      acc[g] = wmma_bf16(Ap, Bv, acc[g]);
    }
  }

  // ---- normalize and emit bf16 for the output projection GEMM ----
#pragma unroll
  for (int r = 0; r < 8; ++r) {
    float inv = 1.0f / l[r];
    int row = q0 + r + half * 8;
    unsigned short* orow = attnout + (b * SS + row) * (HH * DK) + h * DK;
#pragma unroll
    for (int g = 0; g < 4; ++g)
      orow[g * 16 + l16] = f2bf(acc[g][r] * inv);
  }
}

// ---------------------------------------------------------------------------
// Host-side launch
// ---------------------------------------------------------------------------
extern "C" void kernel_launch(void* const* d_in, const int* in_sizes, int n_in,
                              void* d_out, int out_size, void* d_ws, size_t ws_size,
                              hipStream_t stream) {
  const float* x    = (const float*)d_in[0];
  const float* Wq   = (const float*)d_in[1];
  const float* Wk   = (const float*)d_in[2];
  const float* Wv   = (const float*)d_in[3];
  const float* Wo   = (const float*)d_in[4];
  const float* rel  = (const float*)d_in[5];
  const float* mask = (const float*)d_in[6];

  char* ws = (char*)d_ws;
  // Workspace layout (bytes)
  unsigned short* Xbf  = (unsigned short*)(ws + 0);          //  8 MB [4096][1024]
  unsigned short* WqT  = (unsigned short*)(ws + 8388608);    //  2 MB [1024][1024]
  unsigned short* WkT  = (unsigned short*)(ws + 10485760);   //  2 MB
  unsigned short* WvT  = (unsigned short*)(ws + 12582912);   //  2 MB
  unsigned short* WoT  = (unsigned short*)(ws + 14680064);   //  2 MB
  unsigned short* Qbuf = (unsigned short*)(ws + 16777216);   //  8 MB [b][h][s][d]
  unsigned short* Kbuf = (unsigned short*)(ws + 25165824);   //  8 MB [b][h][s][d]
  unsigned short* Vt   = (unsigned short*)(ws + 33554432);   //  8 MB [b][h][d][s]
  unsigned short* Attn = (unsigned short*)(ws + 41943040);   //  8 MB [b*s][h*d]
  float*          bTab = (float*)(ws + 50331648);            // 262 KB [16][4095]

  float* outp  = (float*)d_out;            // [2,2048,1024]
  float* pbias = outp + (BB * SS * DD);    // [16,2048,2048]

  // 1) precision conversion + weight transposes
  cvt_bf16_kernel<<<(MROWS * DD + 255) / 256, 256, 0, stream>>>(x, Xbf, MROWS * DD);
  transpose_cvt_kernel<<<(DD * DD) / 256, 256, 0, stream>>>(Wq, WqT);
  transpose_cvt_kernel<<<(DD * DD) / 256, 256, 0, stream>>>(Wk, WkT);
  transpose_cvt_kernel<<<(DD * DD) / 256, 256, 0, stream>>>(Wv, WvT);
  transpose_cvt_kernel<<<(DD * DD) / 256, 256, 0, stream>>>(Wo, WoT);

  // 2) distance->bias table, then stream out the 268MB position_bias output
  biastab_kernel<<<(HH * NBIAS + 255) / 256, 256, 0, stream>>>(rel, bTab);
  posbias_kernel<<<(HH * SS * SS / 4 + 255) / 256, 256, 0, stream>>>(bTab, pbias);

  // 3) QKV projections (bf16 WMMA, ping-pong pipelined)
  gemm_bf16_kernel<<<1024, 128, 0, stream>>>(Xbf, WqT, (void*)Qbuf, 0);
  gemm_bf16_kernel<<<1024, 128, 0, stream>>>(Xbf, WkT, (void*)Kbuf, 1);
  gemm_bf16_kernel<<<1024, 128, 0, stream>>>(Xbf, WvT, (void*)Vt,   2);

  // 4) flash attention (async K/V staging, bias+mask fused, WMMA both GEMMs)
  attn_kernel<<<BB * HH * (SS / 64), 128, 0, stream>>>(Qbuf, Kbuf, Vt, bTab, mask, Attn);

  // 5) output projection -> fp32 result
  gemm_bf16_kernel<<<1024, 128, 0, stream>>>(Attn, WoT, (void*)outp, 3);
}
